// KernelSelfAttention_31061203484849
// MI455X (gfx1250) — matrix operand order repaired and checked by reading, not compiled, not run on hardware
//
#include <hip/hip_runtime.h>
#include <cmath>

// Problem constants (match reference)
constexpr int S_  = 2048;
constexpr int H_  = 768;
constexpr int NH_ = 12;
constexpr int HD_ = 64;
constexpr int NF_ = 128;

// Fused attention kernel geometry
constexpr int NW_       = 8;              // waves per block
constexpr int COLS_PW_  = S_ / NW_;       // 256 columns per wave
constexpr int LDSTRIDE_ = S_ + 4;         // padded LDS row stride (floats)

typedef __attribute__((ext_vector_type(16))) _Float16 v16h;
typedef __attribute__((ext_vector_type(8)))  float    v8f;

// ---------------------------------------------------------------------------
// WMMA lane-layout helpers (CDNA5 ISA 7.12.2, wave32)
// ---------------------------------------------------------------------------

// A-matrix 16x32 f16 tile. base points at element (m0, k0) of a row-major
// [*, ld] f16 array. Lane L: row = L&15, K-half = (L>>4)*8.
__device__ inline v16h load_A16(const _Float16* __restrict__ base, int ld, int lane) {
  const int row = lane & 15;
  const int kh  = (lane >> 4) * 8;
  const _Float16* p = base + (size_t)row * ld + kh;
  v16h a;
#pragma unroll
  for (int i = 0; i < 8; ++i) { a[i] = p[i]; a[8 + i] = p[16 + i]; }
  return a;
}

// B-matrix 32x16 f16 tile where B[k][n] = base[n*ld + k].
__device__ inline v16h load_B16(const _Float16* __restrict__ base, int ld, int lane) {
  const int col = lane & 15;
  const int kh  = (lane >> 4) * 8;
  const _Float16* p = base + (size_t)col * ld + kh;
  v16h b;
#pragma unroll
  for (int i = 0; i < 8; ++i) { b[i] = p[i]; b[8 + i] = p[16 + i]; }
  return b;
}

__device__ inline v8f wmma_f16(v16h a, v16h b, v8f c) {
  return __builtin_amdgcn_wmma_f32_16x16x32_f16(
      /*neg_a=*/false, a, /*neg_b=*/false, b,
      /*c_mod=*/(short)0, c, /*reuse_a=*/false, /*reuse_b=*/false);
}

// ---------------------------------------------------------------------------
// f32 -> f16 conversion
// ---------------------------------------------------------------------------
__global__ void cvt_f32_to_f16(const float* __restrict__ src,
                               _Float16* __restrict__ dst, int n) {
  int i = blockIdx.x * blockDim.x + threadIdx.x;
  if (i < n) dst[i] = (_Float16)src[i];
}

// ---------------------------------------------------------------------------
// Y[M,N] = X16[M,K] @ W16[N,K]^T + bias   (one wave computes a 16x64 tile)
// mode 0: write f32 Yf[M,N] and f16 Y16[M,N]        (Q, K projections)
// mode 1: write f16 transposed Y16[col*M + row]     (V -> (NH,HD,S) layout)
// mode 2: write f32 Yf only                          (output projection)
// grid = (M/16, N/64), block = 32
// ---------------------------------------------------------------------------
__global__ void gemm_bT_kernel(const _Float16* __restrict__ X,
                               const _Float16* __restrict__ W,
                               const float* __restrict__ bias,
                               float* __restrict__ Yf,
                               _Float16* __restrict__ Y16,
                               int M, int N, int K, int mode) {
  const int lane = threadIdx.x;
  const int m0 = blockIdx.x * 16;
  const int n0 = blockIdx.y * 64;

  v8f acc[4] = {};
  for (int k0 = 0; k0 < K; k0 += 32) {
    v16h a = load_A16(X + (size_t)m0 * K + k0, K, lane);
#pragma unroll
    for (int nt = 0; nt < 4; ++nt) {
      v16h b = load_B16(W + (size_t)(n0 + nt * 16) * K + k0, K, lane);
      acc[nt] = wmma_f16(a, b, acc[nt]);
    }
  }

  const int colb = lane & 15;
  const int rh   = (lane >> 4) * 8;
#pragma unroll
  for (int nt = 0; nt < 4; ++nt) {
    const int col = n0 + nt * 16 + colb;
    const float bv = bias ? bias[col] : 0.0f;
#pragma unroll
    for (int j = 0; j < 8; ++j) {
      const int row = m0 + rh + j;
      const float val = acc[nt][j] + bv;
      if (mode == 1) {
        Y16[(size_t)col * M + row] = (_Float16)val;       // head-transposed V
      } else {
        Yf[(size_t)row * N + col] = val;
        if (mode == 0) Y16[(size_t)row * N + col] = (_Float16)val;
      }
    }
  }
}

// ---------------------------------------------------------------------------
// RFF feature map: phi = normalize(0.125 * cos(qn . omega^T + b))
// grid = (NH*S, 2) [y=0 -> Q, y=1 -> K], block = NF_ (128)
// ---------------------------------------------------------------------------
__global__ void phi_kernel(const float* __restrict__ Qf,
                           const float* __restrict__ Kf,
                           const float* __restrict__ omega,
                           const float* __restrict__ rffb,
                           _Float16* __restrict__ phiQ,
                           _Float16* __restrict__ phiK) {
  const int idx = blockIdx.x;           // h*S + s
  const int h = idx / S_;
  const int s = idx % S_;
  const bool isK = (blockIdx.y != 0);
  const float* x = (isK ? Kf : Qf) + (size_t)s * H_ + h * HD_;
  _Float16* out  = (isK ? phiK : phiQ) + ((size_t)h * S_ + s) * NF_;

  __shared__ float xn[HD_];
  __shared__ float red[NF_];
  const int tid = threadIdx.x;

  float v = (tid < HD_) ? x[tid] : 0.0f;
  red[tid] = v * v;
  __syncthreads();
  for (int off = NF_ / 2; off > 0; off >>= 1) {
    if (tid < off) red[tid] += red[tid + off];
    __syncthreads();
  }
  const float qn = sqrtf(red[0]) + 1e-5f;   // EPS_NORM
  __syncthreads();
  if (tid < HD_) xn[tid] = v / qn;
  __syncthreads();

  float z = 0.0f;
  const float* om = omega + ((size_t)h * NF_ + tid) * HD_;
#pragma unroll 8
  for (int d = 0; d < HD_; ++d) z += xn[d] * om[d];
  z = z + rffb[h * NF_ + tid];              // SIGMA == 1.0
  const float ph = 0.125f * cosf(z);        // sqrt(2/128) == 0.125 exactly

  red[tid] = ph * ph;
  __syncthreads();
  for (int off = NF_ / 2; off > 0; off >>= 1) {
    if (tid < off) red[tid] += red[tid + off];
    __syncthreads();
  }
  const float pn = sqrtf(red[0]) + 1e-6f;
  out[tid] = (_Float16)(ph / pn);
}

// ---------------------------------------------------------------------------
// Fused scores + softmax + P@V. One block owns 16 rows of one head; the full
// 16 x 2048 f32 score stripe lives in LDS (320KB/WGP makes this possible).
// The weights array is written to HBM exactly once (final probabilities).
// grid = (S/16, NH), block = 32*NW_, dynamic LDS.
// ---------------------------------------------------------------------------
__global__ void fused_attn_kernel(const _Float16* __restrict__ Q16,
                                  const _Float16* __restrict__ K16,
                                  const _Float16* __restrict__ phiQ,
                                  const _Float16* __restrict__ phiK,
                                  const float* __restrict__ mask,
                                  const _Float16* __restrict__ Vt,   // (NH*HD, S)
                                  float* __restrict__ Wt,            // (NH, S, S)
                                  _Float16* __restrict__ ctx16) {    // (S, H)
  extern __shared__ float smem[];
  float* sc     = smem;                       // 16 x LDSTRIDE_ scores
  float* ctxred = sc + 16 * LDSTRIDE_;        // 16 x 64 context reduction
  float* rmax   = ctxred + 16 * 64;           // 16
  float* rsum   = rmax + 16;                  // 16
  float* pred   = rsum + 16;                  // 256 partial reduction scratch

  const int tid  = threadIdx.x;
  const int lane = tid & 31;
  const int wav  = tid >> 5;
  const int s0   = blockIdx.x * 16;
  const int h    = blockIdx.y;

  const float DOT_SCALE = 0.9f / 8.0f;        // ALPHA / sqrt(HD)
  const float KER_SCALE = 1.0f - 0.9f;        // 1 - ALPHA

  // Zero the context reduction buffer.
  for (int i = tid; i < 16 * 64; i += 32 * NW_) ctxred[i] = 0.0f;

  // ---- Phase 1: raw scores for this wave's 256-column slice -> LDS ----
  v16h aq0 = load_A16(Q16 + (size_t)s0 * H_ + h * HD_ + 0,  H_, lane);
  v16h aq1 = load_A16(Q16 + (size_t)s0 * H_ + h * HD_ + 32, H_, lane);
  v16h ap[4];
#pragma unroll
  for (int k = 0; k < 4; ++k)
    ap[k] = load_A16(phiQ + ((size_t)h * S_ + s0) * NF_ + 32 * k, NF_, lane);

  const int colb = lane & 15;
  const int rh   = (lane >> 4) * 8;
  const int cbase = wav * COLS_PW_;

  for (int g = 0; g < COLS_PW_; g += 64) {
#pragma unroll
    for (int nt = 0; nt < 4; ++nt) {
      const int tb = cbase + g + nt * 16;
      v8f cd = {};
      cd = wmma_f16(aq0, load_B16(K16 + (size_t)tb * H_ + h * HD_ + 0,  H_, lane), cd);
      cd = wmma_f16(aq1, load_B16(K16 + (size_t)tb * H_ + h * HD_ + 32, H_, lane), cd);
      v8f cp = {};
#pragma unroll
      for (int k = 0; k < 4; ++k)
        cp = wmma_f16(ap[k], load_B16(phiK + ((size_t)h * S_ + tb) * NF_ + 32 * k, NF_, lane), cp);

      const int col = tb + colb;
      const float mterm = (mask[col] - 1.0f) * 10000.0f;
#pragma unroll
      for (int j = 0; j < 8; ++j)
        sc[(rh + j) * LDSTRIDE_ + col] = DOT_SCALE * cd[j] + KER_SCALE * cp[j] + mterm;
    }
  }
  __syncthreads();

  // ---- Phase 2: per-row max then sum(exp(s-max)) over 2048 columns ----
  {
    const int r  = tid & 15;                  // row
    const int ch = tid >> 4;                  // 16 chunks of 128 columns
    const float* rowp = sc + r * LDSTRIDE_ + ch * 128;
    float mx = -3.0e38f;
    for (int i = 0; i < 128; ++i) mx = fmaxf(mx, rowp[i]);
    pred[r * 16 + ch] = mx;
    __syncthreads();
    if (tid < 16) {
      float m = -3.0e38f;
      for (int c = 0; c < 16; ++c) m = fmaxf(m, pred[tid * 16 + c]);
      rmax[tid] = m;
    }
    __syncthreads();
    const float m = rmax[r];
    float sm = 0.0f;
    for (int i = 0; i < 128; ++i) sm += __expf(rowp[i] - m);
    pred[r * 16 + ch] = sm;
    __syncthreads();
    if (tid < 16) {
      float s = 0.0f;
      for (int c = 0; c < 16; ++c) s += pred[tid * 16 + c];
      rsum[tid] = s;
    }
    __syncthreads();
  }

  // ---- Phase 3: final weights -> HBM (single write) + partial P@V ----
  {
    const int row   = lane & 15;
    const int khalf = lane >> 4;
    const float mx  = rmax[row];
    const float inv = 1.0f / rsum[row];
    const float* srow = sc + row * LDSTRIDE_;
    float* wrow = Wt + ((size_t)h * S_ + s0 + row) * S_;

    v8f acc[4] = {};
    for (int t0 = cbase; t0 < cbase + COLS_PW_; t0 += 32) {
      v16h a;
#pragma unroll
      for (int j = 0; j < 8; ++j) {
        const int c0 = t0 + khalf * 8 + j;
        const float p0 = __expf(srow[c0] - mx) * inv;
        wrow[c0] = p0;
        a[j] = (_Float16)p0;
        const int c1 = t0 + 16 + khalf * 8 + j;
        const float p1 = __expf(srow[c1] - mx) * inv;
        wrow[c1] = p1;
        a[8 + j] = (_Float16)p1;
      }
#pragma unroll
      for (int nt = 0; nt < 4; ++nt) {
        v16h b = load_B16(Vt + ((size_t)(h * HD_ + nt * 16)) * S_ + t0, S_, lane);
        acc[nt] = wmma_f16(a, b, acc[nt]);
      }
    }
    // Accumulate partial context across waves via LDS float atomics.
#pragma unroll
    for (int nt = 0; nt < 4; ++nt) {
      const int d = nt * 16 + colb;
#pragma unroll
      for (int j = 0; j < 8; ++j)
        atomicAdd(&ctxred[(rh + j) * 64 + d], acc[nt][j]);
    }
  }
  __syncthreads();

  // ---- Phase 4: emit context as f16 ----
  for (int i = tid; i < 16 * 64; i += 32 * NW_) {
    const int r = i >> 6;
    const int d = i & 63;
    ctx16[(size_t)(s0 + r) * H_ + h * HD_ + d] = (_Float16)ctxred[i];
  }
}

// ---------------------------------------------------------------------------
// Host launch
// ---------------------------------------------------------------------------
extern "C" void kernel_launch(void* const* d_in, const int* in_sizes, int n_in,
                              void* d_out, int out_size, void* d_ws, size_t ws_size,
                              hipStream_t stream) {
  (void)in_sizes; (void)n_in; (void)out_size; (void)ws_size;

  const float* hidden = (const float*)d_in[0];
  const float* mask   = (const float*)d_in[1];
  const float* Wq     = (const float*)d_in[2];
  const float* bq     = (const float*)d_in[3];
  const float* Wk     = (const float*)d_in[4];
  const float* bk     = (const float*)d_in[5];
  const float* Wv     = (const float*)d_in[6];
  const float* bv     = (const float*)d_in[7];
  const float* Wo     = (const float*)d_in[8];
  const float* bo     = (const float*)d_in[9];
  const float* omega  = (const float*)d_in[10];
  const float* rffb   = (const float*)d_in[11];

  float* out_f   = (float*)d_out;                       // (1, S, H)
  float* weights = out_f + (size_t)S_ * H_;             // (1, NH, S, S)

  // Workspace carve-out (256B aligned)
  char* base = (char*)d_ws;
  size_t off = 0;
  auto take = [&](size_t bytes) -> char* {
    off = (off + 255) & ~(size_t)255;
    char* p = base + off;
    off += bytes;
    return p;
  };
  _Float16* hid16  = (_Float16*)take((size_t)S_ * H_ * 2);
  _Float16* Wq16   = (_Float16*)take((size_t)H_ * H_ * 2);
  _Float16* Wk16   = (_Float16*)take((size_t)H_ * H_ * 2);
  _Float16* Wv16   = (_Float16*)take((size_t)H_ * H_ * 2);
  _Float16* Wo16   = (_Float16*)take((size_t)H_ * H_ * 2);
  float*    Qf     = (float*)take((size_t)S_ * H_ * 4);
  float*    Kf     = (float*)take((size_t)S_ * H_ * 4);
  _Float16* Q16    = (_Float16*)take((size_t)S_ * H_ * 2);
  _Float16* K16    = (_Float16*)take((size_t)S_ * H_ * 2);
  _Float16* Vt16   = (_Float16*)take((size_t)H_ * S_ * 2);      // (NH*HD, S)
  _Float16* phiQ16 = (_Float16*)take((size_t)NH_ * S_ * NF_ * 2);
  _Float16* phiK16 = (_Float16*)take((size_t)NH_ * S_ * NF_ * 2);
  _Float16* ctx16  = (_Float16*)take((size_t)S_ * H_ * 2);

  // 1) f32 -> f16 conversions
  {
    int n = S_ * H_;
    cvt_f32_to_f16<<<(n + 255) / 256, 256, 0, stream>>>(hidden, hid16, n);
    n = H_ * H_;
    cvt_f32_to_f16<<<(n + 255) / 256, 256, 0, stream>>>(Wq, Wq16, n);
    cvt_f32_to_f16<<<(n + 255) / 256, 256, 0, stream>>>(Wk, Wk16, n);
    cvt_f32_to_f16<<<(n + 255) / 256, 256, 0, stream>>>(Wv, Wv16, n);
    cvt_f32_to_f16<<<(n + 255) / 256, 256, 0, stream>>>(Wo, Wo16, n);
  }

  // 2) Q, K, V projections (WMMA)
  dim3 ggrid(S_ / 16, H_ / 64);
  gemm_bT_kernel<<<ggrid, 32, 0, stream>>>(hid16, Wq16, bq, Qf, Q16,
                                           S_, H_, H_, /*mode=*/0);
  gemm_bT_kernel<<<ggrid, 32, 0, stream>>>(hid16, Wk16, bk, Kf, K16,
                                           S_, H_, H_, /*mode=*/0);
  gemm_bT_kernel<<<ggrid, 32, 0, stream>>>(hid16, Wv16, bv, nullptr, Vt16,
                                           S_, H_, H_, /*mode=*/1);

  // 3) RFF features
  phi_kernel<<<dim3(NH_ * S_, 2), NF_, 0, stream>>>(Qf, Kf, omega, rffb,
                                                    phiQ16, phiK16);

  // 4) Fused scores + softmax + P@V; weights written to HBM exactly once.
  {
    const size_t smem_bytes =
        (size_t)(16 * LDSTRIDE_ + 16 * 64 + 16 + 16 + 256) * sizeof(float);
    fused_attn_kernel<<<dim3(S_ / 16, NH_), 32 * NW_, smem_bytes, stream>>>(
        Q16, K16, phiQ16, phiK16, mask, Vt16, weights, ctx16);
  }

  // 5) Output projection (WMMA) -> d_out output region
  gemm_bT_kernel<<<ggrid, 32, 0, stream>>>(ctx16, Wo16, bo, out_f, nullptr,
                                           S_, H_, H_, /*mode=*/2);
}
